// MHA_4750233829670
// MI455X (gfx1250) — compile-verified, matrix operand order checked by reading, not run on hardware
//
#include <hip/hip_runtime.h>
#include <hip/hip_bf16.h>
#include <math.h>

// ---------------------------------------------------------------------------
// Problem constants (from the reference)
// ---------------------------------------------------------------------------
constexpr int Bn    = 16;
constexpr int LQ    = 2048;
constexpr int EMB1  = 1024;
constexpr int EMB2  = 768;
constexpr int HDIM  = 1024;
constexpr int ROWS  = Bn * LQ;      // 32768 flattened Q rows
constexpr int LDS_P = 40;           // bf16/row: 80B pitch, 16B aligned, 20-bank
                                    // step -> conflict-free ds_load_b128
constexpr int KC    = 32;           // k-chunk (one WMMA K depth)
constexpr int NIT   = EMB1 / KC;    // 32 k-steps
constexpr int MT    = 128;          // WG rows   (4 waves x 32 rows)
constexpr int NT    = 128;          // WG cols   (2 waves x 64 cols = 2 heads)

typedef __attribute__((ext_vector_type(16))) __bf16 v16bf;
typedef __attribute__((ext_vector_type(8)))  __bf16 v8bf;
typedef __attribute__((ext_vector_type(8)))  float  v8f;

union frag { v16bf v; struct { v8bf lo, hi; } h; };

__device__ __forceinline__ void load_frag(frag& f, const __bf16* p) {
    f.h.lo = *(const v8bf*)(p);        // K 0..7   (8..15 in upper half-wave)
    f.h.hi = *(const v8bf*)(p + 16);   // K 16..23 (24..31)
}

// per-thread staged global data for one k-chunk (A:128x32, B:128x32 fp32)
struct Stage { float4 a[4]; float4 b[4]; };

__device__ __forceinline__ void stage_load(Stage& s,
    const float* __restrict__ emb1, const float* __restrict__ Wq,
    int rowbase, int colblk, int kk, int tid)
{
    #pragma unroll
    for (int t = 0; t < 4; ++t) {          // A: 128 rows x 32 k = 1024 float4
        int idx = tid + t * 256, r = idx >> 3, kq = idx & 7;
        s.a[t] = *(const float4*)(emb1 + (size_t)(rowbase + r) * EMB1 + kk + kq * 4);
    }
    #pragma unroll
    for (int t = 0; t < 4; ++t) {          // B: 128 n x 32 k = 1024 float4
        int idx = tid + t * 256, n = idx >> 3, kq = idx & 7;
        s.b[t] = *(const float4*)(Wq + (size_t)(colblk + n) * EMB1 + kk + kq * 4);
    }
}

__device__ __forceinline__ void stage_store(__bf16* Ab, __bf16* Bb,
                                            const Stage& s, int tid)
{
    #pragma unroll
    for (int t = 0; t < 4; ++t) {
        int idx = tid + t * 256, r = idx >> 3, kq = idx & 7;
        __bf16* d = &Ab[r * LDS_P + kq * 4];
        d[0] = (__bf16)s.a[t].x; d[1] = (__bf16)s.a[t].y;
        d[2] = (__bf16)s.a[t].z; d[3] = (__bf16)s.a[t].w;
    }
    #pragma unroll
    for (int t = 0; t < 4; ++t) {
        int idx = tid + t * 256, n = idx >> 3, kq = idx & 7;
        __bf16* d = &Bb[n * LDS_P + kq * 4];
        d[0] = (__bf16)s.b[t].x; d[1] = (__bf16)s.b[t].y;
        d[2] = (__bf16)s.b[t].z; d[3] = (__bf16)s.b[t].w;
    }
}

// ---------------------------------------------------------------------------
// Kernel 1: K = emb2 @ Wk^T + bk ; V = emb2 @ Wv^T + bv   (tiny: 16x1024 each)
// ---------------------------------------------------------------------------
__global__ __launch_bounds__(256) void kv_kernel(
    const float* __restrict__ emb2,
    const float* __restrict__ Wk, const float* __restrict__ bk,
    const float* __restrict__ Wv, const float* __restrict__ bv,
    float* __restrict__ Kv, float* __restrict__ Vv)
{
    int id   = blockIdx.x * 256 + threadIdx.x;   // 0 .. 2*16*1024-1
    int kind = id >> 14;                         // 0 = K, 1 = V
    int rem  = id & 16383;
    int b    = rem >> 10;
    int j    = rem & 1023;
    const float* W    = kind ? Wv : Wk;
    const float* bias = kind ? bv : bk;
    const float* e2   = emb2 + b * EMB2;
    const float* w    = W + (size_t)j * EMB2;
    float acc = 0.f;
    #pragma unroll 8
    for (int e = 0; e < EMB2; ++e) acc += w[e] * e2[e];
    acc += bias[j];
    (kind ? Vv : Kv)[b * HDIM + j] = acc;
}

// ---------------------------------------------------------------------------
// Kernel 2: fused Q-GEMM (bf16 WMMA, f32 accum) + per-head sigmoid gate * V
//   grid = (ROWS/128, HDIM/128), block = 256 (8 waves)
//   wave (waveM, waveN): 32 rows x 64 cols -> 8 WMMA tiles, B-frags reused 2x
//   LDS double-buffered; next chunk's global loads overlap current compute
// ---------------------------------------------------------------------------
__global__ __launch_bounds__(256) void fused_qgemm_gate(
    const float* __restrict__ emb1, const float* __restrict__ Wq,
    const float* __restrict__ bq,
    const float* __restrict__ Kv,  const float* __restrict__ Vv,
    float* __restrict__ out)
{
    __shared__ __attribute__((aligned(16))) __bf16 Alds[2][MT * LDS_P];
    __shared__ __attribute__((aligned(16))) __bf16 Blds[2][NT * LDS_P];

    const int tid   = threadIdx.x;
    const int lane  = tid & 31;
    const int wid   = tid >> 5;       // 0..7
    const int waveM = wid & 3;        // 32-row group within 128-row tile
    const int waveN = wid >> 2;       // 64-col group within 128-col tile

    const int rowbase = blockIdx.x * MT;          // flattened (b*LQ + q)
    const int colblk  = blockIdx.y * NT;
    const int colbase = colblk + waveN * 64;      // head-aligned (dh = 64)

    const int lrow  = lane & 15;
    const int khalf = lane >> 4;      // ISA 16-bit A/B layout half-wave K split

    v8f a00 = {}, a01 = {}, a02 = {}, a03 = {};   // row group 0, col tiles 0..3
    v8f a10 = {}, a11 = {}, a12 = {}, a13 = {};   // row group 1

    // ---- prologue: stage chunk 0 into buffer 0
    Stage st;
    stage_load(st, emb1, Wq, rowbase, colblk, 0, tid);
    stage_store(Alds[0], Blds[0], st, tid);

    #pragma unroll 2
    for (int it = 0; it < NIT; ++it) {
        __syncthreads();              // buffer it&1 ready; other buffer free

        const bool more = (it + 1) < NIT;
        if (more)                     // issue next chunk's global loads early
            stage_load(st, emb1, Wq, rowbase, colblk, (it + 1) * KC, tid);

        // ---- all 12 fragment loads up front, then 8 back-to-back WMMAs
        const __bf16* Ab = Alds[it & 1];
        const __bf16* Bb = Blds[it & 1];
        frag ua0, ua1, ub0, ub1, ub2, ub3;
        load_frag(ua0, &Ab[(waveM * 32      + lrow) * LDS_P + khalf * 8]);
        load_frag(ua1, &Ab[(waveM * 32 + 16 + lrow) * LDS_P + khalf * 8]);
        load_frag(ub0, &Bb[(waveN * 64 +  0 + lrow) * LDS_P + khalf * 8]);
        load_frag(ub1, &Bb[(waveN * 64 + 16 + lrow) * LDS_P + khalf * 8]);
        load_frag(ub2, &Bb[(waveN * 64 + 32 + lrow) * LDS_P + khalf * 8]);
        load_frag(ub3, &Bb[(waveN * 64 + 48 + lrow) * LDS_P + khalf * 8]);

        a00 = __builtin_amdgcn_wmma_f32_16x16x32_bf16(false, ua0.v, false, ub0.v,
                                                      (short)0, a00, false, false);
        a01 = __builtin_amdgcn_wmma_f32_16x16x32_bf16(false, ua0.v, false, ub1.v,
                                                      (short)0, a01, false, false);
        a02 = __builtin_amdgcn_wmma_f32_16x16x32_bf16(false, ua0.v, false, ub2.v,
                                                      (short)0, a02, false, false);
        a03 = __builtin_amdgcn_wmma_f32_16x16x32_bf16(false, ua0.v, false, ub3.v,
                                                      (short)0, a03, false, false);
        a10 = __builtin_amdgcn_wmma_f32_16x16x32_bf16(false, ua1.v, false, ub0.v,
                                                      (short)0, a10, false, false);
        a11 = __builtin_amdgcn_wmma_f32_16x16x32_bf16(false, ua1.v, false, ub1.v,
                                                      (short)0, a11, false, false);
        a12 = __builtin_amdgcn_wmma_f32_16x16x32_bf16(false, ua1.v, false, ub2.v,
                                                      (short)0, a12, false, false);
        a13 = __builtin_amdgcn_wmma_f32_16x16x32_bf16(false, ua1.v, false, ub3.v,
                                                      (short)0, a13, false, false);

        if (more)                     // write next chunk into the other buffer
            stage_store(Alds[(it + 1) & 1], Blds[(it + 1) & 1], st, tid);
    }

    // ---- epilogue: att[m] = <Q[m, head], K[b, head]>, out = sigmoid(att) * V
    const int b = rowbase / LQ;       // 128-row tile never crosses a batch
    float bqv[4], kvv[4], vvv[4];
    #pragma unroll
    for (int t = 0; t < 4; ++t) {
        int c  = colbase + t * 16 + lrow;
        bqv[t] = bq[c];
        kvv[t] = Kv[b * HDIM + c];
        vvv[t] = Vv[b * HDIM + c];
    }

    #pragma unroll
    for (int rg = 0; rg < 2; ++rg) {
        const v8f& c0 = rg ? a10 : a00;
        const v8f& c1 = rg ? a11 : a01;
        const v8f& c2 = rg ? a12 : a02;
        const v8f& c3 = rg ? a13 : a03;
        #pragma unroll
        for (int i = 0; i < 8; ++i) {
            // C/D layout: VGPR i, lanes 0-15 -> M=i, lanes 16-31 -> M=i+8
            float part = (c0[i] + bqv[0]) * kvv[0]
                       + (c1[i] + bqv[1]) * kvv[1]
                       + (c2[i] + bqv[2]) * kvv[2]
                       + (c3[i] + bqv[3]) * kvv[3];
            #pragma unroll
            for (int m = 1; m < 16; m <<= 1)   // reduce within 16-lane half-wave
                part += __shfl_xor(part, m, 16);
            float s = 1.0f / (1.0f + __expf(-part));
            int row = rowbase + waveM * 32 + rg * 16 + i + 8 * khalf;
            float* o = out + (size_t)row * HDIM + colbase + lrow;
            o[0]  = s * vvv[0];
            o[16] = s * vvv[1];
            o[32] = s * vvv[2];
            o[48] = s * vvv[3];
        }
    }
}

// ---------------------------------------------------------------------------
// Launch
// ---------------------------------------------------------------------------
extern "C" void kernel_launch(void* const* d_in, const int* in_sizes, int n_in,
                              void* d_out, int out_size, void* d_ws, size_t ws_size,
                              hipStream_t stream) {
    const float* emb1 = (const float*)d_in[0];
    const float* emb2 = (const float*)d_in[1];
    const float* Wq   = (const float*)d_in[2];
    const float* bq   = (const float*)d_in[3];
    const float* Wk   = (const float*)d_in[4];
    const float* bk   = (const float*)d_in[5];
    const float* Wv   = (const float*)d_in[6];
    const float* bv   = (const float*)d_in[7];
    // d_in[8] = n_head scalar (fixed to 16 in this specialization)
    float* out = (float*)d_out;

    float* Kv = (float*)d_ws;            // 16*1024 f32
    float* Vv = Kv + Bn * HDIM;          // 16*1024 f32  (128 KB total scratch)

    kv_kernel<<<dim3((2 * Bn * HDIM) / 256), dim3(256), 0, stream>>>(
        emb2, Wk, bk, Wv, bv, Kv, Vv);

    fused_qgemm_gate<<<dim3(ROWS / MT, HDIM / NT), dim3(256), 0, stream>>>(
        emb1, Wq, bq, Kv, Vv, out);
}
